// LlamaAttention_86612310491459
// MI455X (gfx1250) — compile-verified, matrix operand order checked by reading, not run on hardware
//
#include <hip/hip_runtime.h>
#include <hip/hip_bf16.h>

typedef __attribute__((ext_vector_type(16))) __bf16 v16bf;
typedef __attribute__((ext_vector_type(8)))  __bf16 v8bf;
typedef __attribute__((ext_vector_type(8)))  float  v8f;

namespace {
constexpr int kB   = 2;
constexpr int kS   = 2048;
constexpr int kH   = 4096;
constexpr int kNQ  = 32;
constexpr int kNKV = 8;
constexpr int kHD  = 128;
constexpr int BM = 128, BN = 128, BK = 32;
constexpr int PAD = 40;    // bf16 elements per LDS row; 80 B => 16-B aligned v8bf slots
}

// Native f32->bf16 (lowers to v_cvt_pk_bf16_f32-class hardware converts on gfx1250,
// instead of the 3-VALU-op software RNE sequence).
__device__ __forceinline__ __bf16 to_bf16(float f) { return (__bf16)f; }

__device__ __forceinline__ v8bf cvt8(v8f f) {
  return __builtin_convertvector(f, v8bf);
}

__device__ __forceinline__ v8f zero8() {
  v8f z;
#pragma unroll
  for (int e = 0; e < 8; ++e) z[e] = 0.0f;
  return z;
}

__device__ __forceinline__ v16bf combine16(v8bf lo, v8bf hi) {
  v16bf r;
#pragma unroll
  for (int e = 0; e < 8; ++e) { r[e] = lo[e]; r[8 + e] = hi[e]; }
  return r;
}

__device__ __forceinline__ v8f wmma_bf16(v16bf a, v16bf b, v8f c) {
  // D = A(16x32 bf16) x B(32x16 bf16) + C(16x16 f32)
  return __builtin_amdgcn_wmma_f32_16x16x32_bf16(false, a, false, b,
                                                 (short)0, c, false, false);
}

// ---------------------------------------------------------------------------
// Generic C[M,N] = A[M,K] * B[K,N], f32 in/out, bf16 WMMA math.
// Block: 256 threads (8 waves), tile 128x128, K-step 32.
// ---------------------------------------------------------------------------
__global__ __launch_bounds__(256, 2)
void gemm_bf16(const float* __restrict__ A, const float* __restrict__ Bw,
               float* __restrict__ C, int M, int N, int K) {
  __shared__ __bf16 sA[BM * PAD];        // [m][k]
  __shared__ __bf16 sB[BN * PAD];        // transposed: [n][k]

  const int tid  = threadIdx.x;
  const int lane = tid & 31;
  const int wave = tid >> 5;
  const int wr = wave >> 1, wc = wave & 1;         // 4x2 wave grid -> 32x64 per wave
  const int m0 = blockIdx.y * BM;
  const int n0 = blockIdx.x * BN;
  const int lrow = lane & 15;
  const int kb   = (lane >> 4) << 3;               // A/B fragment K-base per half-wave
  const int rtop = (lane >> 4) << 3;               // C fragment row offset

  const int arow = tid >> 1, acol = (tid & 1) << 4;   // 2 thr/row, 16 floats each
  const int brow = tid >> 3, bcol = (tid & 7) << 4;   // 8 thr/row, 16 floats each

  v8f acc[2][4];
#pragma unroll
  for (int i = 0; i < 2; ++i)
#pragma unroll
    for (int j = 0; j < 4; ++j) acc[i][j] = zero8();

  for (int k0 = 0; k0 < K; k0 += BK) {
    {   // A slab: 128x32 f32 -> bf16 LDS (row-major)
      const float* ag = A + (size_t)(m0 + arow) * K + (k0 + acol);
      float t[16];
      *(float4*)&t[0]  = ((const float4*)ag)[0];
      *(float4*)&t[4]  = ((const float4*)ag)[1];
      *(float4*)&t[8]  = ((const float4*)ag)[2];
      *(float4*)&t[12] = ((const float4*)ag)[3];
      *(v8bf*)&sA[arow * PAD + acol]     = cvt8(*(const v8f*)&t[0]);
      *(v8bf*)&sA[arow * PAD + acol + 8] = cvt8(*(const v8f*)&t[8]);
      if (k0 + BK < K) __builtin_prefetch(ag + BK, 0, 3);
    }
    {   // B slab: 32x128 f32 -> bf16 LDS transposed [n][k]
      const float* bg = Bw + (size_t)(k0 + brow) * N + (n0 + bcol);
      float t[16];
      *(float4*)&t[0]  = ((const float4*)bg)[0];
      *(float4*)&t[4]  = ((const float4*)bg)[1];
      *(float4*)&t[8]  = ((const float4*)bg)[2];
      *(float4*)&t[12] = ((const float4*)bg)[3];
#pragma unroll
      for (int e = 0; e < 16; ++e) sB[(bcol + e) * PAD + brow] = to_bf16(t[e]);
      if (k0 + BK < K) __builtin_prefetch(bg + (size_t)BK * N, 0, 3);
    }
    __syncthreads();

    v16bf af[2], bfr[4];
#pragma unroll
    for (int i = 0; i < 2; ++i) {
      const int r = wr * 32 + i * 16 + lrow;
      af[i] = combine16(*(const v8bf*)&sA[r * PAD + kb],
                        *(const v8bf*)&sA[r * PAD + kb + 16]);
    }
#pragma unroll
    for (int j = 0; j < 4; ++j) {
      const int c = wc * 64 + j * 16 + lrow;
      bfr[j] = combine16(*(const v8bf*)&sB[c * PAD + kb],
                         *(const v8bf*)&sB[c * PAD + kb + 16]);
    }
#pragma unroll
    for (int i = 0; i < 2; ++i)
#pragma unroll
      for (int j = 0; j < 4; ++j)
        acc[i][j] = wmma_bf16(af[i], bfr[j], acc[i][j]);
    __syncthreads();
  }

#pragma unroll
  for (int i = 0; i < 2; ++i)
#pragma unroll
    for (int j = 0; j < 4; ++j) {
      const int gcol = n0 + wc * 64 + j * 16 + lrow;
#pragma unroll
      for (int r = 0; r < 8; ++r) {
        const int grow = m0 + wr * 32 + i * 16 + rtop + r;
        C[(size_t)grow * N + gcol] = acc[i][j][r];
      }
    }
}

// ---------------------------------------------------------------------------
// RoPE (rotate-half), in place on [B*S, nheads, 128] f32. One thread per
// (token, head, freq-index i in [0,64)).
// ---------------------------------------------------------------------------
__global__ __launch_bounds__(256)
void rope_kernel(float* __restrict__ x, const int* __restrict__ pos,
                 int nheads, int total) {
  const int idx = blockIdx.x * blockDim.x + threadIdx.x;
  if (idx >= total) return;
  const int i   = idx & 63;
  const int hh  = (idx >> 6) % nheads;
  const int tok = idx / (64 * nheads);
  const int b = tok / kS, srow = tok % kS;
  const float p = (float)pos[b * kS + srow];
  const float f = p * __powf(10000.0f, -(2.0f * (float)i) / (float)kHD);
  float s, c;
  __sincosf(f, &s, &c);
  float* px = x + ((size_t)tok * nheads + hh) * kHD;
  const float x1 = px[i], x2 = px[i + 64];
  px[i]      = x1 * c - x2 * s;
  px[i + 64] = x2 * c + x1 * s;
}

// ---------------------------------------------------------------------------
// Flash attention forward (causal, GQA 32q/8kv, HD=128), bf16 WMMA.
// Grid: (B*NQ, S/64). Block: 128 threads = 4 waves, 16 query rows per wave.
// ---------------------------------------------------------------------------
__global__ __launch_bounds__(128, 2)
void flash_fwd(const float* __restrict__ q, const float* __restrict__ kk,
               const float* __restrict__ vv, const int* __restrict__ amask,
               float* __restrict__ o) {
  __shared__ __bf16 sK[32 * 136];          // K tile  [key][hd], padded
  __shared__ __bf16 sVt[kHD * PAD];        // V tile transposed [hd][key]
  __shared__ __bf16 sP[4][16 * PAD];       // per-wave P staging [row][key]

  const int tid  = threadIdx.x;
  const int lane = tid & 31;
  const int wave = tid >> 5;
  const int b   = blockIdx.x / kNQ;
  const int h   = blockIdx.x % kNQ;
  const int kvh = h / (kNQ / kNKV);
  const int qt  = blockIdx.y;
  const int qbase = qt * 64 + wave * 16;
  const int lrow = lane & 15;
  const int kb   = (lane >> 4) << 3;
  const int rtop = (lane >> 4) << 3;
  const float sm_scale = 0.088388347648318447f;   // 1/sqrt(128), folded into Q

  // Q fragments (16 rows x 128, K-chunks of 32), scaled, kept in registers.
  v16bf qf[4];
  {
    const float* qr = q + ((size_t)(b * kS + qbase + lrow) * kNQ + h) * kHD;
#pragma unroll
    for (int c = 0; c < 4; ++c) {
      float t[16];
      *(float4*)&t[0]  = *(const float4*)(qr + c * 32 + kb);
      *(float4*)&t[4]  = *(const float4*)(qr + c * 32 + kb + 4);
      *(float4*)&t[8]  = *(const float4*)(qr + c * 32 + kb + 16);
      *(float4*)&t[12] = *(const float4*)(qr + c * 32 + kb + 20);
      v8f lo = *(const v8f*)&t[0] * sm_scale;
      v8f hi = *(const v8f*)&t[8] * sm_scale;
      qf[c] = combine16(cvt8(lo), cvt8(hi));
    }
  }

  float mrow[8], lsum[8];
  v8f accO[8];
#pragma unroll
  for (int r = 0; r < 8; ++r) { mrow[r] = -__builtin_inff(); lsum[r] = 0.0f; }
#pragma unroll
  for (int j = 0; j < 8; ++j) accO[j] = zero8();

  const int keyk = tid >> 2;            // 0..31
  const int segk = (tid & 3) << 5;      // 0,32,64,96
  const int nkt  = qt * 2 + 2;          // block-uniform causal bound

  for (int kt = 0; kt < nkt; ++kt) {
    {   // cooperative K/V tile load: 32 keys x 128, f32 -> bf16
      const size_t base = ((size_t)(b * kS + kt * 32 + keyk) * kNKV + kvh) * kHD + segk;
      const float* kg = kk + base;
      const float* vg = vv + base;
      float tk[32], tv[32];
#pragma unroll
      for (int e4 = 0; e4 < 8; ++e4) {
        *(float4*)&tk[e4 * 4] = *(const float4*)(kg + e4 * 4);
        *(float4*)&tv[e4 * 4] = *(const float4*)(vg + e4 * 4);
      }
#pragma unroll
      for (int e8 = 0; e8 < 4; ++e8)
        *(v8bf*)&sK[keyk * 136 + segk + e8 * 8] = cvt8(*(const v8f*)&tk[e8 * 8]);
#pragma unroll
      for (int e = 0; e < 32; ++e) sVt[(segk + e) * PAD + keyk] = to_bf16(tv[e]);
      if (kt + 1 < nkt) {
        __builtin_prefetch(kg + (size_t)32 * kNKV * kHD, 0, 3);
        __builtin_prefetch(vg + (size_t)32 * kNKV * kHD, 0, 3);
      }
    }
    __syncthreads();

    // S = Q * K^T : two 16x16 score tiles (keys 0-15 / 16-31), 8 WMMAs
    v8f s0 = zero8(), s1 = zero8();
#pragma unroll
    for (int c = 0; c < 4; ++c) {
      v16bf kf0 = combine16(*(const v8bf*)&sK[lrow * 136 + c * 32 + kb],
                            *(const v8bf*)&sK[lrow * 136 + c * 32 + kb + 16]);
      v16bf kf1 = combine16(*(const v8bf*)&sK[(16 + lrow) * 136 + c * 32 + kb],
                            *(const v8bf*)&sK[(16 + lrow) * 136 + c * 32 + kb + 16]);
      s0 = wmma_bf16(qf[c], kf0, s0);
      s1 = wmma_bf16(qf[c], kf1, s1);
    }

    // mask + online softmax (row = rtop + r, replicated over a 16-lane half)
#pragma unroll
    for (int r = 0; r < 8; ++r) {
      const int qg  = qbase + rtop + r;
      const int kg0 = kt * 32 + lrow;
      const int kg1 = kg0 + 16;
      float sv0 = s0[r], sv1 = s1[r];
      if (kg0 > qg || amask[b * kS + kg0] == 0) sv0 = -__builtin_inff();
      if (kg1 > qg || amask[b * kS + kg1] == 0) sv1 = -__builtin_inff();
      float mx = fmaxf(sv0, sv1);
#pragma unroll
      for (int off = 8; off; off >>= 1) mx = fmaxf(mx, __shfl_xor(mx, off, 32));
      const float mnew  = fmaxf(mrow[r], mx);
      const float alpha = __expf(mrow[r] - mnew);
      const float p0 = __expf(sv0 - mnew);
      const float p1 = __expf(sv1 - mnew);
      float rs = p0 + p1;
#pragma unroll
      for (int off = 8; off; off >>= 1) rs += __shfl_xor(rs, off, 32);
      lsum[r] = lsum[r] * alpha + rs;
      mrow[r] = mnew;
#pragma unroll
      for (int j = 0; j < 8; ++j) accO[j][r] *= alpha;
      sP[wave][(rtop + r) * PAD + lrow]      = to_bf16(p0);
      sP[wave][(rtop + r) * PAD + 16 + lrow] = to_bf16(p1);
    }
    // wave-private LDS RAW: drain DS counter before reloading P as A-fragment
    asm volatile("s_wait_dscnt 0" ::: "memory");

    // O += P(16x32) * V(32x128) : 8 WMMAs
    v16bf pf = combine16(*(const v8bf*)&sP[wave][lrow * PAD + kb],
                         *(const v8bf*)&sP[wave][lrow * PAD + kb + 16]);
#pragma unroll
    for (int j = 0; j < 8; ++j) {
      v16bf vf = combine16(*(const v8bf*)&sVt[(j * 16 + lrow) * PAD + kb],
                           *(const v8bf*)&sVt[(j * 16 + lrow) * PAD + kb + 16]);
      accO[j] = wmma_bf16(pf, vf, accO[j]);
    }
    __syncthreads();
  }

#pragma unroll
  for (int j = 0; j < 8; ++j)
#pragma unroll
    for (int r = 0; r < 8; ++r) {
      const int qg = qbase + rtop + r;
      o[((size_t)(b * kS + qg) * kNQ + h) * kHD + j * 16 + lrow] =
          accO[j][r] / lsum[r];
    }
}

// ---------------------------------------------------------------------------
extern "C" void kernel_launch(void* const* d_in, const int* in_sizes, int n_in,
                              void* d_out, int out_size, void* d_ws, size_t ws_size,
                              hipStream_t stream) {
  (void)in_sizes; (void)n_in; (void)out_size; (void)ws_size;

  const float* hs    = (const float*)d_in[0];
  const float* Wq    = (const float*)d_in[1];
  const float* Wk    = (const float*)d_in[2];
  const float* Wv    = (const float*)d_in[3];
  const float* Wo    = (const float*)d_in[4];
  const int*   amask = (const int*)d_in[5];
  const int*   pos   = (const int*)d_in[6];

  float* qb = (float*)d_ws;                               // [B,S,NQ,HD]
  float* kb = qb + (size_t)kB * kS * kNQ * kHD;           // [B,S,NKV,HD]
  float* vb = kb + (size_t)kB * kS * kNKV * kHD;          // [B,S,NKV,HD]
  float* ob = vb + (size_t)kB * kS * kNKV * kHD;          // [B,S,NQ,HD]

  const int M = kB * kS;   // 4096 tokens
  dim3 blk(256);

  // QKV projections
  gemm_bf16<<<dim3((kNQ  * kHD) / BN, M / BM), blk, 0, stream>>>(hs, Wq, qb, M, kNQ  * kHD, kH);
  gemm_bf16<<<dim3((kNKV * kHD) / BN, M / BM), blk, 0, stream>>>(hs, Wk, kb, M, kNKV * kHD, kH);
  gemm_bf16<<<dim3((kNKV * kHD) / BN, M / BM), blk, 0, stream>>>(hs, Wv, vb, M, kNKV * kHD, kH);

  // RoPE on Q and K
  const int tq = M * kNQ  * 64;
  const int tk = M * kNKV * 64;
  rope_kernel<<<(tq + 255) / 256, blk, 0, stream>>>(qb, pos, kNQ,  tq);
  rope_kernel<<<(tk + 255) / 256, blk, 0, stream>>>(kb, pos, kNKV, tk);

  // Causal GQA flash attention
  flash_fwd<<<dim3(kB * kNQ, kS / 64), dim3(128), 0, stream>>>(qb, kb, vb, amask, ob);

  // Output projection into d_out
  gemm_bf16<<<dim3(kH / BN, M / BM), blk, 0, stream>>>(ob, Wo, (float*)d_out, M, kH, kH);
}
